// GraphAttentionLayer_10977936408769
// MI455X (gfx1250) — matrix operand order repaired and checked by reading, not compiled, not run on hardware
//
#include <hip/hip_runtime.h>

#define NN 8192   // nodes
#define DD 256    // input dim
#define FF 128    // out channels
#define BB 4096   // lookup batch

typedef _Float16 v16h __attribute__((ext_vector_type(16)));
typedef float    v8f  __attribute__((ext_vector_type(8)));

union Frag16 {
  v16h  h;
  float4 f4[2];   // 32 bytes = 16 halves
};

__device__ __forceinline__ v8f wmma_f16(v16h a, v16h b, v8f c) {
  // D = A(16x32 f16) * B(32x16 f16) + C(16x16 f32)
  return __builtin_amdgcn_wmma_f32_16x16x32_f16(
      /*neg_a=*/false, a, /*neg_b=*/false, b,
      /*c_mod=*/(short)0, c, /*reuse_a=*/false, /*reuse_b=*/false);
}

// ---------------------------------------------------------------------------
// Kernel 1: Wh = emb @ W in full f32, rounded once to f16 for the WMMA stages.
// ---------------------------------------------------------------------------
__global__ void __launch_bounds__(256)
wh_gemm_kernel(const float* __restrict__ emb, const float* __restrict__ W,
               _Float16* __restrict__ WhH) {
  const int i = blockIdx.x * 2 + (threadIdx.x >> 7);
  const int f = threadIdx.x & 127;
  const float* er = emb + (size_t)i * DD;
  float acc = 0.0f;
#pragma unroll 8
  for (int k = 0; k < DD; ++k)
    acc = fmaf(er[k], W[k * FF + f], acc);
  WhH[(size_t)i * FF + f] = (_Float16)acc;
}

// ---------------------------------------------------------------------------
// Kernel 2: fused masked attention (flash-style online softmax) + ELU.
// 64-j panels; double-buffered B fragments under the WMMA chains; row sums
// via P@ones WMMA; O rescale only when the running max advances (wave vote).
// adj (256 MB) is streamed exactly once.
// ---------------------------------------------------------------------------
__global__ void __launch_bounds__(128)
gat_flash_kernel(const _Float16* __restrict__ WhH, const int* __restrict__ adj,
                 float* __restrict__ Helu) {
  __shared__ alignas(16) _Float16 Tnf[64 * 128];    // K-panel, [j][f]  16 KB
  __shared__ alignas(16) _Float16 Tfn[128 * 64];    // V-panel, [f][j]  16 KB
  __shared__ alignas(16) _Float16 Pbuf[4][16 * 64]; // per-wave P transpose 8 KB

  const int tid  = threadIdx.x;
  const int wave = tid >> 5;
  const int lane = tid & 31;
  const int lg   = lane & 15;   // group lane (N / M index)
  const int hh   = lane >> 4;   // lane half

  const int i0 = blockIdx.x * 64 + wave * 16;

  // Q fragments: A-layout, lane: M = lg, K = hh*16 + 0..15
  Frag16 qa[4];
#pragma unroll
  for (int kk = 0; kk < 4; ++kk) {
    const float4* p =
        (const float4*)(WhH + (size_t)(i0 + lg) * FF + kk * 32 + hh * 16);
    qa[kk].f4[0] = p[0];   // K = kk*32 + hh*16 + 0..7
    qa[kk].f4[1] = p[1];   // K = kk*32 + hh*16 + 8..15
  }

  // all-ones B fragment for the row-sum WMMA
  Frag16 ones;
#pragma unroll
  for (int e = 0; e < 16; ++e) ones.h[e] = (_Float16)1.0f;

  v8f zero = {};
  v8f o[8];
#pragma unroll
  for (int n = 0; n < 8; ++n) o[n] = zero;
  float mrow[8], lrow[8];
#pragma unroll
  for (int r = 0; r < 8; ++r) { mrow[r] = -9.0e15f; lrow[r] = 0.0f; }

  for (int jb = 0; jb < NN / 64; ++jb) {
    const int j0 = jb * 64;
    __syncthreads();
    // --- stage the 64x128 f16 j-panel in both majors; rows paired so the
    //     transposed copy stores packed b32 ---
#pragma unroll
    for (int c = 0; c < 4; ++c) {
      int idx = tid + c * 128;        // 512 row-pair chunks
      int j   = (idx >> 4) << 1;      // 0,2,..,62
      int fc  = (idx & 15) << 3;
      float4 v0 = *(const float4*)(WhH + (size_t)(j0 + j) * FF + fc);
      float4 v1 = *(const float4*)(WhH + (size_t)(j0 + j + 1) * FF + fc);
      *(float4*)(&Tnf[j * 128 + fc]) = v0;
      *(float4*)(&Tnf[(j + 1) * 128 + fc]) = v1;
      union { float4 f; _Float16 e[8]; } u0, u1;
      u0.f = v0; u1.f = v1;
#pragma unroll
      for (int q = 0; q < 8; ++q) {
        union { _Float16 h[2]; unsigned u; } pk;
        pk.h[0] = u0.e[q]; pk.h[1] = u1.e[q];
        *(unsigned*)&Tfn[(fc + q) * 64 + j] = pk.u;
      }
    }
    __syncthreads();

    // --- S = Q @ K^T : four 16x16 f32 tiles, double-buffered B fragments.
    //     Order kk-outer/t-inner: same accumulator WMMAs are 4 apart. ---
    v8f s[4];
#pragma unroll
    for (int t = 0; t < 4; ++t) s[t] = zero;
    {
      Frag16 sb[2];
      const float4* p0 = (const float4*)(&Tnf[lg * 128 + hh * 16]);
      sb[0].f4[0] = p0[0]; sb[0].f4[1] = p0[1];
#pragma unroll
      for (int it = 0; it < 16; ++it) {
        const int kk = it >> 2, t = it & 3;
        if (it + 1 < 16) {
          const int kn = (it + 1) >> 2, tn = (it + 1) & 3;
          const float4* pn =
              (const float4*)(&Tnf[(tn * 16 + lg) * 128 + kn * 32 + hh * 16]);
          sb[(it + 1) & 1].f4[0] = pn[0];
          sb[(it + 1) & 1].f4[1] = pn[1];
        }
        s[t] = wmma_f16(qa[kk].h, sb[it & 1].h, s[t]);
      }
    }

    // --- mask with adj (C layout: VGPR r -> M = r + 8*hh, lane -> N) ---
#pragma unroll
    for (int t = 0; t < 4; ++t) {
#pragma unroll
      for (int r = 0; r < 8; ++r) {
        int m = adj[(size_t)(i0 + r + 8 * hh) * NN + (j0 + t * 16 + lg)];
        s[t][r] = (m > 0) ? s[t][r] : -9.0e15f;
      }
    }
    if (jb + 1 < NN / 64) {    // uniform branch
#pragma unroll
      for (int r = 0; r < 8; ++r)
        __builtin_prefetch(&adj[(size_t)(i0 + r + 8 * hh) * NN + j0 + 64 + lg],
                           0, 0);
    }

    // --- running max update (only cross-lane reduction left) ---
    float mold[8], mnew[8];
    int chg = 0;
#pragma unroll
    for (int r = 0; r < 8; ++r) {
      float v = fmaxf(fmaxf(s[0][r], s[1][r]), fmaxf(s[2][r], s[3][r]));
      v = fmaxf(v, __shfl_xor(v, 1, 32));
      v = fmaxf(v, __shfl_xor(v, 2, 32));
      v = fmaxf(v, __shfl_xor(v, 4, 32));
      v = fmaxf(v, __shfl_xor(v, 8, 32));
      mold[r] = mrow[r];
      mnew[r] = fmaxf(mrow[r], v);
      chg |= (mnew[r] > mold[r]) ? 1 : 0;
      mrow[r] = mnew[r];
    }
    // rescale O and l only when some row max advanced (rare). No WMMA inside,
    // so EXEC masking here is safe even if the branch is not scalarized.
    if (__any(chg)) {
      float scl[8];
#pragma unroll
      for (int r = 0; r < 8; ++r) {
        scl[r] = __expf(mold[r] - mnew[r]);  // finite sentinel -> no NaNs
        lrow[r] *= scl[r];
      }
#pragma unroll
      for (int n = 0; n < 8; ++n)
#pragma unroll
        for (int r = 0; r < 8; ++r) o[n][r] *= scl[r];
    }

    // --- exponentiate in place ---
#pragma unroll
    for (int t = 0; t < 4; ++t)
#pragma unroll
      for (int r = 0; r < 8; ++r) s[t][r] = __expf(s[t][r] - mnew[r]);

    // --- P: C-fragment -> A-fragment transpose via per-wave LDS scratch ---
    _Float16* pb = &Pbuf[wave][0];
#pragma unroll
    for (int t = 0; t < 4; ++t)
#pragma unroll
      for (int r = 0; r < 8; ++r)
        pb[(r + 8 * hh) * 64 + t * 16 + lg] = (_Float16)s[t][r];
    __builtin_amdgcn_wave_barrier();
    asm volatile("s_wait_dscnt 0" ::: "memory");
    Frag16 pa[2];
#pragma unroll
    for (int kk = 0; kk < 2; ++kk) {
      pa[kk].f4[0] = *(const float4*)(&pb[lg * 64 + kk * 32 + hh * 16]);
      pa[kk].f4[1] = *(const float4*)(&pb[lg * 64 + kk * 32 + hh * 16 + 8]);
    }

    // --- row sums of P via WMMA against ones (every N column == rowsum) ---
    v8f d = zero;
    d = wmma_f16(pa[0].h, ones.h, d);
    d = wmma_f16(pa[1].h, ones.h, d);

    // --- O += P @ V : double-buffered V fragments.
    //     Order kk-outer/n-inner: same accumulator WMMAs are 8 apart. ---
    {
      Frag16 vb[2];
      const float4* v0 = (const float4*)(&Tfn[lg * 64 + hh * 16]);
      vb[0].f4[0] = v0[0]; vb[0].f4[1] = v0[1];
#pragma unroll
      for (int it = 0; it < 16; ++it) {
        const int kk = it >> 3, n = it & 7;
        if (it + 1 < 16) {
          const int kn = (it + 1) >> 3, nn = (it + 1) & 7;
          const float4* vp =
              (const float4*)(&Tfn[(nn * 16 + lg) * 64 + kn * 32 + hh * 16]);
          vb[(it + 1) & 1].f4[0] = vp[0];
          vb[(it + 1) & 1].f4[1] = vp[1];
        }
        o[n] = wmma_f16(pa[kk].h, vb[it & 1].h, o[n]);
      }
    }
#pragma unroll
    for (int r = 0; r < 8; ++r) lrow[r] += d[r];
  }

  // --- epilogue: normalize by l, ELU, store f32 ---
#pragma unroll
  for (int n = 0; n < 8; ++n) {
#pragma unroll
    for (int r = 0; r < 8; ++r) {
      float v = o[n][r] / lrow[r];
      v = (v > 0.0f) ? v : expm1f(v);
      Helu[(size_t)(i0 + r + 8 * hh) * FF + n * 16 + lg] = v;
    }
  }
}

// ---------------------------------------------------------------------------
// Kernel 3: out = normalize(Helu[x])  (one wave per output row)
// ---------------------------------------------------------------------------
__global__ void __launch_bounds__(256)
lookup_norm_kernel(const int* __restrict__ x, const float* __restrict__ H,
                   float* __restrict__ out) {
  const int row  = blockIdx.x * 8 + (threadIdx.x >> 5);
  const int lane = threadIdx.x & 31;
  const int src  = x[row];
  float4 v = *(const float4*)(H + (size_t)src * FF + lane * 4);
  float s = v.x * v.x + v.y * v.y + v.z * v.z + v.w * v.w;
#pragma unroll
  for (int d = 1; d < 32; d <<= 1) s += __shfl_xor(s, d, 32);
  float inv = 1.0f / fmaxf(sqrtf(s), 1e-12f);
  float4 oo; oo.x = v.x * inv; oo.y = v.y * inv; oo.z = v.z * inv; oo.w = v.w * inv;
  *(float4*)(out + (size_t)row * FF + lane * 4) = oo;
}

extern "C" void kernel_launch(void* const* d_in, const int* in_sizes, int n_in,
                              void* d_out, int out_size, void* d_ws,
                              size_t ws_size, hipStream_t stream) {
  (void)in_sizes; (void)n_in; (void)out_size; (void)ws_size;
  const int*   x   = (const int*)d_in[0];
  const int*   adj = (const int*)d_in[1];
  const float* emb = (const float*)d_in[2];
  const float* W   = (const float*)d_in[3];
  float* out = (float*)d_out;

  _Float16* WhH  = (_Float16*)d_ws;                                   // 2 MB
  float*    Helu = (float*)((char*)d_ws + (size_t)NN * FF * sizeof(_Float16)); // 4 MB

  wh_gemm_kernel<<<NN / 2, 256, 0, stream>>>(emb, W, WhH);
  gat_flash_kernel<<<NN / 64, 128, 0, stream>>>(WhH, adj, Helu);
  lookup_norm_kernel<<<BB / 8, 256, 0, stream>>>(x, Helu, out);
}
